// BaseLayers_35459249995852
// MI455X (gfx1250) — matrix operand order repaired and checked by reading, not compiled, not run on hardware
//
#include <hip/hip_runtime.h>
#include <hip/hip_bf16.h>
#include <stdint.h>

typedef __attribute__((ext_vector_type(16))) _Float16 v16h;
typedef __attribute__((ext_vector_type(8)))  _Float16 v8h;
typedef __attribute__((ext_vector_type(8)))  float    v8f;

// ---------------------------------------------------------------------------
// Pass 1: per-(dst,rel) edge counts + per-relation histogram (LDS partials)
// ---------------------------------------------------------------------------
__global__ void count_kernel(const int* __restrict__ ei, const int* __restrict__ et,
                             int* __restrict__ cnt_i, int* __restrict__ relCnt,
                             int E, int R) {
    __shared__ int lh[64];
    if (threadIdx.x < 64) lh[threadIdx.x] = 0;
    __syncthreads();
    int e = blockIdx.x * blockDim.x + threadIdx.x;
    if (e < E) {
        int dst = ei[E + e];
        int rel = et[e];
        atomicAdd(&cnt_i[(size_t)dst * R + rel], 1);
        atomicAdd(&lh[rel], 1);
    }
    __syncthreads();
    if (threadIdx.x < R) {
        int c = lh[threadIdx.x];
        if (c) atomicAdd(&relCnt[threadIdx.x], c);
    }
}

// relOff (exclusive prefix of relCnt), per-relation 16-edge group starts
__global__ void prefix_kernel(const int* __restrict__ relCnt, int* __restrict__ relOff,
                              int* __restrict__ groupStart, int* __restrict__ totalG, int R) {
    if (blockIdx.x == 0 && threadIdx.x == 0) {
        int ro = 0, gs = 0;
        for (int r = 0; r < R; ++r) {
            relOff[r] = ro; groupStart[r] = gs;
            ro += relCnt[r];
            gs += (relCnt[r] + 15) >> 4;
        }
        relOff[R] = ro; groupStart[R] = gs; totalG[0] = gs;
    }
}

// Counting-sort scatter of edge ids into relation bins (block-local LDS batching)
__global__ void scatter_kernel(const int* __restrict__ et, const int* __restrict__ relOff,
                               int* __restrict__ relCursor, int* __restrict__ binned,
                               int E, int R) {
    __shared__ int lcnt[64];
    __shared__ int lbase[64];
    if (threadIdx.x < 64) lcnt[threadIdx.x] = 0;
    __syncthreads();
    int e = blockIdx.x * blockDim.x + threadIdx.x;
    bool ok = e < E;
    int rel = 0, lpos = 0;
    if (ok) { rel = et[e]; lpos = atomicAdd(&lcnt[rel], 1); }
    __syncthreads();
    if (threadIdx.x < R) {
        int c = lcnt[threadIdx.x];
        lbase[threadIdx.x] = c ? atomicAdd(&relCursor[threadIdx.x], c) : 0;
    }
    __syncthreads();
    if (ok) binned[relOff[rel] + lbase[rel] + lpos] = e;
}

__global__ void inv_kernel(const int* __restrict__ cnt_i, float* __restrict__ invc, int NR) {
    int i = blockIdx.x * blockDim.x + threadIdx.x;
    if (i < NR) {
        int c = cnt_i[i];
        invc[i] = c ? 1.0f / (float)c : 0.0f;
    }
}

// acc1 = root1 + bias1
__global__ void init1_kernel(const float* __restrict__ root1, const float* __restrict__ bias1,
                             float* __restrict__ acc1, int NH, int H) {
    int i = blockIdx.x * blockDim.x + threadIdx.x;
    if (i < NH) acc1[i] = root1[i] + bias1[i % H];
}

// ---------------------------------------------------------------------------
// Layer 1: one wave per edge; acc1[dst,:] += W1[rel,src,:] / cnt[dst,rel]
// ---------------------------------------------------------------------------
__global__ void layer1_kernel(const int* __restrict__ ei, const int* __restrict__ et,
                              const float* __restrict__ W1, const float* __restrict__ invc,
                              float* __restrict__ acc1, int N, int R, int H, int E) {
    int wib  = threadIdx.x >> 5;
    int lane = threadIdx.x & 31;
    int e = blockIdx.x * (blockDim.x >> 5) + wib;
    if (e >= E) return;                      // wave-uniform exit
    int src = ei[e], dst = ei[E + e], rel = et[e];
    float scl = invc[(size_t)dst * R + rel];
    const float4* wrow = (const float4*)(W1 + ((size_t)rel * N + src) * H);
    float* orow = acc1 + (size_t)dst * H;
    for (int h4 = lane; h4 * 4 < H; h4 += 32) {
        float4 v = wrow[h4];
        atomicAdd(&orow[h4 * 4 + 0], v.x * scl);
        atomicAdd(&orow[h4 * 4 + 1], v.y * scl);
        atomicAdd(&orow[h4 * 4 + 2], v.z * scl);
        atomicAdd(&orow[h4 * 4 + 3], v.w * scl);
    }
}

// h16 = (f16) relu(acc1)
__global__ void relu_kernel(const float* __restrict__ acc1, _Float16* __restrict__ h16, int NH) {
    int i = blockIdx.x * blockDim.x + threadIdx.x;
    if (i < NH) {
        float x = acc1[i];
        h16[i] = (_Float16)(x > 0.0f ? x : 0.0f);
    }
}

// ---------------------------------------------------------------------------
// Pack W2[r] (and root2 as slot r==R) into the WMMA B-operand lane layout:
// halves[i] of lane L = B[kb + (i&7) + (i>>3)*16][ L%16 ],  kb = (L<16 ? 0 : 8), per (kc,nt) tile
// ---------------------------------------------------------------------------
template <int HH, int LLd>
__global__ void packw2_kernel(const float* __restrict__ W2, const float* __restrict__ root2,
                              _Float16* __restrict__ w2pk, int R) {
    constexpr int KC = HH / 32, NT = LLd / 16;
    int idx = blockIdx.x * blockDim.x + threadIdx.x;
    int total = (R + 1) * KC * NT * 512;
    if (idx >= total) return;
    int i    = idx & 15;
    int lane = (idx >> 4) & 31;
    int t    = idx >> 9;                 // tile index
    int nt   = t % NT;
    int kc   = (t / NT) % KC;
    int r    = t / (NT * KC);
    int n = nt * 16 + (lane & 15);
    int k = kc * 32 + ((lane < 16) ? 0 : 8) + (i & 7) + ((i >> 3) << 4);
    float v = (r < R) ? W2[((size_t)r * HH + k) * LLd + n]
                      : root2[(size_t)k * LLd + n];
    w2pk[idx] = (_Float16)v;
}

// ---------------------------------------------------------------------------
// Dense WMMA GEMM: out2acc = bias2 + h16 @ root2   (B slot r == R)
// one wave -> 16 rows x 64 cols
// ---------------------------------------------------------------------------
template <int HH, int LLd>
__global__ void root2_gemm_kernel(const _Float16* __restrict__ h16,
                                  const _Float16* __restrict__ w2pk,
                                  const float* __restrict__ bias2,
                                  float* __restrict__ out2acc, int N, int R) {
    constexpr int KC = HH / 32, NT = LLd / 16;
    int lane = threadIdx.x;
    int row0 = blockIdx.x * 16;
    int m    = lane & 15;
    int row  = row0 + m; if (row >= N) row = N - 1;
    v8f zero = {};
    v8f acc[NT];
#pragma unroll
    for (int nt = 0; nt < NT; ++nt) acc[nt] = zero;
    const int kb = (lane < 16) ? 0 : 8;
#pragma unroll
    for (int kc = 0; kc < KC; ++kc) {
        const _Float16* ap = h16 + (size_t)row * HH + kc * 32 + kb;
        v8h lo = *(const v8h*)(ap);
        v8h hi = *(const v8h*)(ap + 16);
        v16h a;
#pragma unroll
        for (int i = 0; i < 8; ++i) { a[i] = lo[i]; a[8 + i] = hi[i]; }
#pragma unroll
        for (int nt = 0; nt < NT; ++nt) {
            v16h b = *(const v16h*)(w2pk + ((((size_t)R * KC + kc) * NT + nt) * 32 + lane) * 16);
            acc[nt] = __builtin_amdgcn_wmma_f32_16x16x32_f16(
                false, a, false, b, (short)0, acc[nt], false, false);
        }
    }
    int col = lane & 15;
#pragma unroll
    for (int v = 0; v < 8; ++v) {
        int wr = row0 + ((lane < 16) ? v : v + 8);
        if (wr < N) {
#pragma unroll
            for (int nt = 0; nt < NT; ++nt)
                out2acc[(size_t)wr * LLd + nt * 16 + col] = bias2[nt * 16 + col] + acc[nt][v];
        }
    }
}

// ---------------------------------------------------------------------------
// Layer-2 einsum: one wave per group of 16 same-relation edges:
// out2acc[dst] += (h16[src] @ W2[rel]) / cnt[dst,rel]   via v_wmma_f32_16x16x32_f16
// ---------------------------------------------------------------------------
template <int HH, int LLd>
__global__ void einsum_edges_kernel(const int* __restrict__ ei, const int* __restrict__ binned,
                                    const int* __restrict__ relOff, const int* __restrict__ groupStart,
                                    const int* __restrict__ totalG, const float* __restrict__ invc,
                                    const _Float16* __restrict__ h16, const _Float16* __restrict__ w2pk,
                                    float* __restrict__ out2acc, int E, int R) {
    constexpr int KC = HH / 32, NT = LLd / 16;
    int g = blockIdx.x;
    if (g >= totalG[0]) return;              // wave-uniform; EXEC stays all-1 for WMMA
    int r = 0;
    for (int rr = R - 1; rr > 0; --rr) { if (g >= groupStart[rr]) { r = rr; break; } }
    int lane = threadIdx.x;
    int m    = lane & 15;
    int base = relOff[r] + (g - groupStart[r]) * 16;
    int end  = relOff[r + 1];
    bool valid = (base + m) < end;
    int e   = valid ? binned[base + m] : 0;
    int src = valid ? ei[e] : 0;
    int dst = valid ? ei[E + e] : 0;
    float scl = valid ? invc[(size_t)dst * R + r] : 0.0f;

    v8f zero = {};
    v8f acc[NT];
#pragma unroll
    for (int nt = 0; nt < NT; ++nt) acc[nt] = zero;

    const int kb = (lane < 16) ? 0 : 8;
#pragma unroll
    for (int kc = 0; kc < KC; ++kc) {
        const _Float16* ap = h16 + (size_t)src * HH + kc * 32 + kb;
        v8h lo = *(const v8h*)(ap);
        v8h hi = *(const v8h*)(ap + 16);
        v16h a;
#pragma unroll
        for (int i = 0; i < 8; ++i) { a[i] = lo[i]; a[8 + i] = hi[i]; }
#pragma unroll
        for (int nt = 0; nt < NT; ++nt) {
            v16h b = *(const v16h*)(w2pk + ((((size_t)r * KC + kc) * NT + nt) * 32 + lane) * 16);
            acc[nt] = __builtin_amdgcn_wmma_f32_16x16x32_f16(
                false, a, false, b, (short)0, acc[nt], false, false);
        }
    }
    // Scatter rows with per-row scale; C/D layout: VGPR v -> row v (lanes 0-15) / v+8
    int col = lane & 15;
#pragma unroll
    for (int v = 0; v < 8; ++v) {
        int rowm = (lane < 16) ? v : v + 8;
        float s = __shfl(scl, rowm, 32);
        int   d = __shfl(dst, rowm, 32);
        if (s != 0.0f) {
#pragma unroll
            for (int nt = 0; nt < NT; ++nt)
                atomicAdd(&out2acc[(size_t)d * LLd + nt * 16 + col], acc[nt][v] * s);
        }
    }
}

__global__ void sigmoid_kernel(const float* __restrict__ out2acc, float* __restrict__ out, int NL) {
    int i = blockIdx.x * blockDim.x + threadIdx.x;
    if (i < NL) {
        float x = out2acc[i];
        out[i] = 1.0f / (1.0f + __expf(-x));
    }
}

// ---------------------------------------------------------------------------
static inline size_t align256(size_t x) { return (x + 255) & ~(size_t)255; }

extern "C" void kernel_launch(void* const* d_in, const int* in_sizes, int n_in,
                              void* d_out, int out_size, void* d_ws, size_t ws_size,
                              hipStream_t stream) {
    const int*   ei    = (const int*)d_in[0];
    const int*   et    = (const int*)d_in[1];
    const float* W1    = (const float*)d_in[2];
    const float* root1 = (const float*)d_in[3];
    const float* bias1 = (const float*)d_in[4];
    const float* W2    = (const float*)d_in[5];
    const float* root2 = (const float*)d_in[6];
    const float* bias2 = (const float*)d_in[7];
    float*       out   = (float*)d_out;

    const int E = in_sizes[1];
    const int H = in_sizes[4];          // 128
    const int L = in_sizes[7];          // 64
    const int N = in_sizes[3] / H;      // 50000
    const int R = in_sizes[5] / (H * L);// 16
    const int NR = N * R, NH = N * H, NL = N * L;
    constexpr int HH = 128, LLd = 64;
    const int KC = HH / 32, NT = LLd / 16;

    // Workspace carve-out
    uint8_t* w = (uint8_t*)d_ws;
    size_t off = 0;
    int*      cnt_i   = (int*)(w + off);      off = align256(off + (size_t)NR * 4);
    float*    invc    = (float*)(w + off);    off = align256(off + (size_t)NR * 4);
    float*    acc1    = (float*)(w + off);    off = align256(off + (size_t)NH * 4);
    _Float16* h16     = (_Float16*)(w + off); off = align256(off + (size_t)NH * 2);
    int*      binned  = (int*)(w + off);      off = align256(off + (size_t)E * 4);
    int*      relCnt  = (int*)(w + off);      // relCnt[16] + relCursor[16] contiguous
    int*      relCursor = relCnt + 16;        off = align256(off + 32 * 4);
    int*      relOff  = (int*)(w + off);      off = align256(off + 17 * 4);
    int*      groupStart = (int*)(w + off);   off = align256(off + 17 * 4);
    int*      totalG  = (int*)(w + off);      off = align256(off + 4);
    _Float16* w2pk    = (_Float16*)(w + off); off = align256(off + (size_t)(R + 1) * KC * NT * 512 * 2);
    float*    out2acc = (float*)(w + off);    off = align256(off + (size_t)NL * 4);

    // Zero the accumulating counters (graph-capture-safe)
    hipMemsetAsync(cnt_i, 0, (size_t)NR * 4, stream);
    hipMemsetAsync(relCnt, 0, 32 * 4, stream);

    const int T = 256;
    // Counts + histogram, prefix, bin edges by relation
    count_kernel<<<(E + T - 1) / T, T, 0, stream>>>(ei, et, cnt_i, relCnt, E, R);
    prefix_kernel<<<1, 32, 0, stream>>>(relCnt, relOff, groupStart, totalG, R);
    scatter_kernel<<<(E + T - 1) / T, T, 0, stream>>>(et, relOff, relCursor, binned, E, R);
    inv_kernel<<<(NR + T - 1) / T, T, 0, stream>>>(cnt_i, invc, NR);

    // Layer 1
    init1_kernel<<<(NH + T - 1) / T, T, 0, stream>>>(root1, bias1, acc1, NH, H);
    layer1_kernel<<<(E + 7) / 8, T, 0, stream>>>(ei, et, W1, invc, acc1, N, R, H, E);
    relu_kernel<<<(NH + T - 1) / T, T, 0, stream>>>(acc1, h16, NH);

    // Pack W2 (+root2 slot) into WMMA B layout
    {
        int total = (R + 1) * KC * NT * 512;
        packw2_kernel<HH, LLd><<<(total + T - 1) / T, T, 0, stream>>>(W2, root2, w2pk, R);
    }

    // Layer 2: dense root2 GEMM writes bias2 + h@root2, then edge einsum atomics
    root2_gemm_kernel<HH, LLd><<<(N + 15) / 16, 32, 0, stream>>>(h16, w2pk, bias2, out2acc, N, R);
    {
        int maxGroups = (E + 15) / 16 + R + 1;   // grid fixed at capture; device-side early exit
        einsum_edges_kernel<HH, LLd><<<maxGroups, 32, 0, stream>>>(
            ei, binned, relOff, groupStart, totalG, invc, h16, w2pk, out2acc, E, R);
    }

    sigmoid_kernel<<<(NL + T - 1) / T, T, 0, stream>>>(out2acc, out, NL);
}
